// Rvt_3917010174405
// MI455X (gfx1250) — matrix operand order repaired and checked by reading, not compiled, not run on hardware
//
#include <hip/hip_runtime.h>
#include <hip/hip_bf16.h>
#include <math.h>

// Problem constants
#define NN 4
#define HH 48
#define WW 48
#define PIX (HH*WW)        // 2304
#define DHH 46
#define DWW 46
#define DD (DHH*DWW)       // 2116
#define CP 16              // channels per head
#define QT (PIX/16)        // 144 q-tiles of 16
#define DB 64              // d-columns per online-softmax block (4 WMMA sub-tiles)
#define NDB ((DD+DB-1)/DB) // 34 blocks (last partial: 4 valid)
#define KTOT 640           // 64 (a) + 576 (im2col x, first 64 ch only)
#define KSTEPS (KTOT/32)   // 20
#define MTILES 12          // 192 stacked gate rows / 16

typedef __attribute__((ext_vector_type(2)))  float  v2f;
typedef __attribute__((ext_vector_type(8)))  float  v8f;
typedef __attribute__((ext_vector_type(16))) __bf16 v16bf;

__device__ __forceinline__ __bf16 f2bf(float f) {
    unsigned u = __builtin_bit_cast(unsigned, f);
    unsigned r = (u + 0x7FFFu + ((u >> 16) & 1u)) >> 16;
    unsigned short s = (unsigned short)r;
    return __builtin_bit_cast(__bf16, s);
}

__device__ __forceinline__ float rmax16(float v) {
    #pragma unroll
    for (int m = 1; m < 16; m <<= 1) v = fmaxf(v, __shfl_xor(v, m, 16));
    return v;
}
__device__ __forceinline__ float rsum16(float v) {
    #pragma unroll
    for (int m = 1; m < 16; m <<= 1) v += __shfl_xor(v, m, 16);
    return v;
}

// ---------------- Stage 1: x = conv1x1(x_in, Wx) + bx  (f32 + bf16 copy) ---
__global__ void proj_kernel(const float* __restrict__ xin,
                            const float* __restrict__ Wx,
                            const float* __restrict__ bx,
                            float* __restrict__ x, __bf16* __restrict__ xb) {
    int e = blockIdx.x * blockDim.x + threadIdx.x;
    if (e >= NN * PIX) return;
    int pix = e % PIX, n = e / PIX;
    float xi[64];
    #pragma unroll 8
    for (int i = 0; i < 64; ++i) xi[i] = xin[(n * 64 + i) * PIX + pix];
    for (int r = 0; r < 64; ++r) {
        float s = bx[r];
        #pragma unroll 8
        for (int i = 0; i < 64; ++i) s += Wx[r * 64 + i] * xi[i];
        x[(n * 64 + r) * PIX + pix]  = s;
        xb[(n * 64 + r) * PIX + pix] = f2bf(s);
    }
}

// ---------------- Stage 2a: q (heads 0,1) grouped 3x3 SAME -----------------
__global__ void conv_q_kernel(const float* __restrict__ x,
                              const float* __restrict__ Wq,
                              float* __restrict__ q) {
    int e = blockIdx.x * blockDim.x + threadIdx.x;
    if (e >= NN * 2 * CP * PIX) return;
    int pix = e % PIX;
    int c   = (e / PIX) % CP;
    int g   = (e / (PIX * CP)) & 1;
    int n   = e / (PIX * CP * 2);
    int oy = pix / WW, ox = pix % WW;
    float s = 0.f;
    for (int ci = 0; ci < 32; ++ci) {
        const float* xc = x  + (n * 64 + g * 32 + ci) * PIX;
        const float* wc = Wq + ((g * CP + c) * 32 + ci) * 9;
        #pragma unroll
        for (int t = 0; t < 9; ++t) {
            int iy = oy + t / 3 - 1, ix = ox + t % 3 - 1;
            if (iy >= 0 && iy < HH && ix >= 0 && ix < WW)
                s += wc[t] * xc[iy * WW + ix];
        }
    }
    q[((n * 2 + g) * CP + c) * PIX + pix] = s;
}

// ---------------- Stage 2b: k,v (heads 0,1) grouped 3x3 VALID --------------
__global__ void conv_kv_kernel(const float* __restrict__ x,
                               const float* __restrict__ Wk,
                               const float* __restrict__ Wv,
                               const float* __restrict__ bv,
                               float* __restrict__ kb, float* __restrict__ vb) {
    int e = blockIdx.x * blockDim.x + threadIdx.x;
    if (e >= NN * 2 * CP * DD) return;
    int d = e % DD;
    int c = (e / DD) % CP;
    int g = (e / (DD * CP)) & 1;
    int n = e / (DD * CP * 2);
    int oy = d / DWW, ox = d % DWW;
    float sk = 0.f, sv = 0.f;
    for (int ci = 0; ci < 32; ++ci) {
        const float* xc  = x  + (n * 64 + g * 32 + ci) * PIX + oy * WW + ox;
        const float* wkc = Wk + ((g * CP + c) * 32 + ci) * 9;
        const float* wvc = Wv + ((g * CP + c) * 32 + ci) * 9;
        #pragma unroll
        for (int t = 0; t < 9; ++t) {
            float xv = xc[(t / 3) * WW + (t % 3)];
            sk += wkc[t] * xv;
            sv += wvc[t] * xv;
        }
    }
    kb[((n * 2 + g) * CP + c) * DD + d] = sk;
    vb[((n * 2 + g) * CP + c) * DD + d] = sv + bv[g * CP + c];
}

// -------- Stage 3: pack stacked gate weights [192 x 640] into bf16 ---------
// in WMMA 16-bit A (16x32) register order: block (mt,ks) of 512 bf16,
// element at lane*16 + vgpr*2 + half.
__global__ void pack_w_kernel(const float* __restrict__ Wia,
                              const float* __restrict__ Wga,
                              const float* __restrict__ Woa,
                              const float* __restrict__ Wix,
                              const float* __restrict__ Wgx,
                              const float* __restrict__ Wox,
                              __bf16* __restrict__ wp) {
    int e = blockIdx.x * blockDim.x + threadIdx.x;
    if (e >= MTILES * KSTEPS * 512) return;
    int blk = e >> 9, within = e & 511;
    int mt = blk / KSTEPS, ks = blk % KSTEPS;
    int lane = within >> 4, elem = within & 15;
    int vg = elem >> 1, hl = elem & 1;
    int m_local = lane & 15;
    int base;
    if (vg < 4) base = (lane < 16 ? 0 : 8) + vg * 2;
    else        base = (lane < 16 ? 16 : 24) + (vg - 4) * 2;
    int k_local = base + hl;
    int row = mt * 16 + m_local;       // 0..191
    int k   = ks * 32 + k_local;       // 0..639
    int gate = row >> 6, r = row & 63;
    const float* W1 = gate == 0 ? Wia : (gate == 1 ? Wga : Woa);
    const float* W3 = gate == 0 ? Wix : (gate == 1 ? Wgx : Wox);
    float val;
    if (k < 64) {
        val = W1[r * 64 + k];
    } else {
        int kk = k - 64, c = kk / 9, t = kk % 9;
        val = W3[(r * 128 + c) * 9 + t];   // h-half (c>=64) of Wx is unused (h==0)
    }
    wp[e] = f2bf(val);
}

// -------- Stage 4: flash attention for heads 0,1, f32 WMMA 16x16x4 ---------
// one wave32 per (n, head, 16-row q tile); 64 d-columns per softmax update so
// cross-lane (ds_bpermute) reductions are amortized 4x vs per-16 tiles.
__global__ __launch_bounds__(32) void attn_kernel(const float* __restrict__ q,
                                                  const float* __restrict__ kb,
                                                  const float* __restrict__ vb,
                                                  __bf16* __restrict__ ab) {
    __shared__ float lP[DB * 16];     // 64 x 16 P block (transposed staging)
    const int lane = threadIdx.x;
    int b  = blockIdx.x;
    int qt = b % QT;
    int g  = (b / QT) & 1;
    int n  = b / (QT * 2);
    const int hf = lane >> 4;      // which 16-lane half
    const int lo = lane & 15;
    const float* qg = q  + ((n * 2 + g) * CP) * PIX;
    const float* kg = kb + ((n * 2 + g) * CP) * DD;
    const float* vg = vb + ((n * 2 + g) * CP) * DD;

    // A = q tile: A[m=qrow, k=c], f32 16x4 layout: m=lane%16, k=j+2*(lane/16)
    v2f A[4];
    #pragma unroll
    for (int s = 0; s < 4; ++s)
        #pragma unroll
        for (int j = 0; j < 2; ++j) {
            int c = 4 * s + j + 2 * hf;
            A[s][j] = qg[c * PIX + qt * 16 + lo];
        }

    v8f acc = {};
    float mrow[8], lrow[8];
    #pragma unroll
    for (int j = 0; j < 8; ++j) { mrow[j] = -1e30f; lrow[j] = 0.f; }

    for (int dt = 0; dt < NDB; ++dt) {
        // logits for 4 sub-tiles of 16 d-columns each
        v8f S[4];
        #pragma unroll
        for (int t = 0; t < 4; ++t) {
            int d = dt * DB + t * 16 + lo; if (d >= DD) d = DD - 1;
            v8f Sv = {};
            #pragma unroll
            for (int s = 0; s < 4; ++s) {
                v2f B;   // B[k=c_sub, n=d]: n=lane%16, k=j+2*(lane/16)
                #pragma unroll
                for (int j = 0; j < 2; ++j) {
                    int c = 4 * s + j + 2 * hf;
                    B[j] = kg[c * DD + d];
                }
                Sv = __builtin_amdgcn_wmma_f32_16x16x4_f32(false, A[s], false, B,
                                                           (short)0, Sv, false, false);
            }
            bool oob = (dt * DB + t * 16 + lo) >= DD;
            #pragma unroll
            for (int j = 0; j < 8; ++j) if (oob) Sv[j] = -1e30f;
            S[t] = Sv;
        }

        // online softmax; row = j + 8*hf. Reduce within-lane over the 4
        // sub-tiles first, then one 16-lane butterfly per row per block.
        #pragma unroll
        for (int j = 0; j < 8; ++j) {
            float tm = fmaxf(fmaxf(S[0][j], S[1][j]), fmaxf(S[2][j], S[3][j]));
            tm = rmax16(tm);
            float nm = fmaxf(mrow[j], tm);
            float alpha = expf(mrow[j] - nm);
            float psum = 0.f;
            #pragma unroll
            for (int t = 0; t < 4; ++t) {
                float p = expf(S[t][j] - nm);
                S[t][j] = p;
                psum += p;
            }
            lrow[j] = lrow[j] * alpha + rsum16(psum);
            mrow[j] = nm;
            acc[j] *= alpha;
        }

        // transpose P block through LDS: C-layout -> A-layout
        __syncthreads();
        #pragma unroll
        for (int t = 0; t < 4; ++t)
            #pragma unroll
            for (int j = 0; j < 8; ++j)
                lP[t * 256 + (j + 8 * hf) * 16 + lo] = S[t][j];
        __syncthreads();

        // acc[q,c] += P[q,d] * V[d,c]  (4 sub-tiles x 4 k-steps)
        #pragma unroll
        for (int t = 0; t < 4; ++t) {
            #pragma unroll
            for (int s = 0; s < 4; ++s) {
                v2f PA, VB;
                #pragma unroll
                for (int j = 0; j < 2; ++j) {
                    int dl = 4 * s + j + 2 * hf;
                    PA[j] = lP[t * 256 + lo * 16 + dl];
                    int d = dt * DB + t * 16 + dl; if (d >= DD) d = DD - 1;
                    VB[j] = vg[lo * DD + d];   // c = lane%16
                }
                acc = __builtin_amdgcn_wmma_f32_16x16x4_f32(false, PA, false, VB,
                                                            (short)0, acc, false, false);
            }
        }
        __syncthreads();
    }
    // acc layout: (qrow = j+8*hf, c = lane%16)
    #pragma unroll
    for (int j = 0; j < 8; ++j) {
        int qpos = qt * 16 + j + 8 * hf;
        ab[(n * 64 + g * CP + lo) * PIX + qpos] = f2bf(acc[j] / lrow[j]);
    }
}

// ------ Stage 5: heads 2,3 of a are the constant bv (softmax uniform) ------
__global__ void fill_a23_kernel(const float* __restrict__ bv,
                                __bf16* __restrict__ ab) {
    int e = blockIdx.x * blockDim.x + threadIdx.x;
    if (e >= NN * 32 * PIX) return;
    int pix = e % PIX;
    int c   = 32 + (e / PIX) % 32;
    int n   = e / (PIX * 32);
    ab[(n * 64 + c) * PIX + pix] = f2bf(bv[c]);
}

// ------ Stage 6: gate GEMM pre[192, pix] = Wpack[192,640] x col[640,pix] ---
// bf16 WMMA 16x16x32; one wave per 16-pixel tile; col built in LDS B-layout
__global__ __launch_bounds__(32) void gates_kernel(const __bf16* __restrict__ wp,
                                                   const __bf16* __restrict__ ab,
                                                   const __bf16* __restrict__ xb,
                                                   float* __restrict__ pre) {
    __shared__ __align__(32) __bf16 lcol[KSTEPS * 512];
    const int lane = threadIdx.x;
    int b = blockIdx.x;
    int tile = b % QT, n = b / QT;
    int pix0 = tile * 16;

    // fill col tile (K=640 x 16 pixels) in WMMA 16-bit B (32x16) order
    for (int e = lane; e < KTOT * 16; e += 32) {
        int k = e >> 4, pl = e & 15;
        int pix = pix0 + pl;
        __bf16 val;
        if (k < 64) {
            val = ab[(n * 64 + k) * PIX + pix];
        } else {
            int kk = k - 64, c = kk / 9, t = kk % 9;
            int py = pix / WW + t / 3 - 1, px = pix % WW + t % 3 - 1;
            if (py >= 0 && py < HH && px >= 0 && px < WW)
                val = xb[(n * 64 + c) * PIX + py * WW + px];
            else
                val = __builtin_bit_cast(__bf16, (unsigned short)0);
        }
        int kr = k & 31;
        int dl = pl + ((kr & 8) << 1);         // lanes 16-31 hold K 8..15, 24..31
        lcol[(k >> 5) * 512 + dl * 16 + (kr & 7) + ((kr >> 1) & 8)] = val;
    }
    __syncthreads();

    v8f acc[MTILES];
    v8f z = {};
    #pragma unroll
    for (int m = 0; m < MTILES; ++m) acc[m] = z;

    for (int ks = 0; ks < KSTEPS; ++ks) {
        v16bf B = *(const v16bf*)(&lcol[ks * 512 + lane * 16]);
        if (ks + 1 < KSTEPS)
            __builtin_prefetch((const void*)(wp + ((ks + 1) * 512 + lane * 16)), 0, 1);
        #pragma unroll
        for (int m = 0; m < MTILES; ++m) {
            v16bf Aw = *(const v16bf*)(wp + ((m * KSTEPS + ks) * 512 + lane * 16));
            acc[m] = __builtin_amdgcn_wmma_f32_16x16x32_bf16(false, Aw, false, B,
                                                             (short)0, acc[m],
                                                             false, false);
        }
    }
    const int hf = lane >> 4, lo = lane & 15;
    #pragma unroll
    for (int m = 0; m < MTILES; ++m)
        #pragma unroll
        for (int j = 0; j < 8; ++j) {
            int row = m * 16 + j + 8 * hf;   // 0..191
            int gate = row >> 6, r = row & 63;
            pre[((n * 3 + gate) * 64 + r) * PIX + pix0 + lo] = acc[m][j];
        }
}

// ------ Stage 7: epilogue  h = sig(po) * tanh( sig(pi) * tanh(pg) ) --------
__global__ void combine_kernel(const float* __restrict__ pre,
                               const float* __restrict__ bi,
                               const float* __restrict__ bg,
                               const float* __restrict__ bo,
                               float* __restrict__ out) {
    int e = blockIdx.x * blockDim.x + threadIdx.x;
    if (e >= NN * 64 * PIX) return;
    int pix = e % PIX;
    int r   = (e / PIX) % 64;
    int n   = e / (PIX * 64);
    float pi = pre[((n * 3 + 0) * 64 + r) * PIX + pix] + bi[r];
    float pg = pre[((n * 3 + 1) * 64 + r) * PIX + pix] + bg[r];
    float po = pre[((n * 3 + 2) * 64 + r) * PIX + pix] + bo[r];
    float gi = 1.f / (1.f + expf(-pi));
    float gg = tanhf(pg);
    float go = 1.f / (1.f + expf(-po));
    out[(n * 64 + r) * PIX + pix] = go * tanhf(gi * gg);   // c was 0 -> f gate dead
}

extern "C" void kernel_launch(void* const* d_in, const int* in_sizes, int n_in,
                              void* d_out, int out_size, void* d_ws, size_t ws_size,
                              hipStream_t stream) {
    (void)in_sizes; (void)n_in; (void)out_size; (void)ws_size;
    const float* x_in = (const float*)d_in[0];
    const float* Wx   = (const float*)d_in[1];
    const float* bx   = (const float*)d_in[2];
    const float* Wq   = (const float*)d_in[3];
    const float* Wk   = (const float*)d_in[4];
    const float* Wv   = (const float*)d_in[5];
    const float* bv   = (const float*)d_in[6];
    const float* Wia  = (const float*)d_in[7];
    const float* Wix  = (const float*)d_in[8];
    const float* bi   = (const float*)d_in[9];
    // d_in[10..12] = Wfa, Wfx, bf : f gate multiplies c==0, mathematically dead
    const float* Wga  = (const float*)d_in[13];
    const float* Wgx  = (const float*)d_in[14];
    const float* bg   = (const float*)d_in[15];
    const float* Woa  = (const float*)d_in[16];
    const float* Wox  = (const float*)d_in[17];
    const float* bo   = (const float*)d_in[18];
    float* out = (float*)d_out;

    char* ws = (char*)d_ws;
    float*  x   = (float*)(ws + 0);              // 4*64*2304 f32
    __bf16* xb  = (__bf16*)(ws + 2359296);       // bf16 copy
    float*  qb  = (float*)(ws + 3538944);        // 4*2*16*2304 f32
    float*  kb  = (float*)(ws + 4718592);        // 4*2*16*2116 f32
    float*  vb  = (float*)(ws + 5801984);        // 4*2*16*2116 f32
    __bf16* ab  = (__bf16*)(ws + 6885376);       // 4*64*2304 bf16
    __bf16* wp  = (__bf16*)(ws + 8065024);       // 192*640 bf16 packed
    float*  pre = (float*)(ws + 8310784);        // 4*3*64*2304 f32

    proj_kernel<<<(NN * PIX + 255) / 256, 256, 0, stream>>>(x_in, Wx, bx, x, xb);
    conv_q_kernel<<<(NN * 2 * CP * PIX + 255) / 256, 256, 0, stream>>>(x, Wq, qb);
    conv_kv_kernel<<<(NN * 2 * CP * DD + 255) / 256, 256, 0, stream>>>(x, Wk, Wv, bv, kb, vb);
    pack_w_kernel<<<(MTILES * KSTEPS * 512 + 255) / 256, 256, 0, stream>>>(
        Wia, Wga, Woa, Wix, Wgx, Wox, wp);
    attn_kernel<<<NN * 2 * QT, 32, 0, stream>>>(qb, kb, vb, ab);
    fill_a23_kernel<<<(NN * 32 * PIX + 255) / 256, 256, 0, stream>>>(bv, ab);
    gates_kernel<<<NN * QT, 32, 0, stream>>>(wp, ab, xb, pre);
    combine_kernel<<<(NN * 64 * PIX + 255) / 256, 256, 0, stream>>>(pre, bi, bg, bo, out);
}
